// CE_73976516706682
// MI455X (gfx1250) — matrix operand order repaired and checked by reading, not compiled, not run on hardware
//
#include <hip/hip_runtime.h>
#include <hip/hip_bf16.h>
#include <stdint.h>

typedef __attribute__((ext_vector_type(16))) __bf16 bf16x16;
typedef __attribute__((ext_vector_type(8)))  __bf16 bf16x8;
typedef __attribute__((ext_vector_type(8)))  float  f32x8;
typedef __attribute__((ext_vector_type(4)))  int    i32x4;

namespace {
constexpr int kBatch = 4;
constexpr int kCin   = 64;
constexpr int kCi    = 16;
constexpr int kH     = 128;
constexpr int kW     = 128;
constexpr int kNPix  = kH * kW;        // 16384
constexpr int kNQ    = 1024;           // 32x32 stride-4 query grid
constexpr int kKs    = 7;
constexpr int kPatch = kCi * kKs * kKs; // 784
constexpr int kKP    = 800;            // 784 padded to mult of 32
constexpr int kFc    = 196;
constexpr int kFcP   = 224;            // 196 padded to mult of 32
constexpr int kPW    = kH + 6;         // 134 (pad 3 each side)
constexpr int kPlane = kPW * kPW;      // 17956
constexpr float kScale = 10.0f;
}

// -------- CDNA5 async global->LDS copy (ASYNCcnt path), with fallback --------
#if defined(__has_builtin)
#if __has_builtin(__builtin_amdgcn_global_load_async_to_lds_b128)
#define GEMM_ASYNC 1
#endif
#endif
#ifndef GEMM_ASYNC
#define GEMM_ASYNC 0
#endif

__device__ __forceinline__ void copy16_g2l(const __bf16* g, __bf16* l) {
#if GEMM_ASYNC
  __builtin_amdgcn_global_load_async_to_lds_b128(
      (__attribute__((address_space(1))) i32x4*)g,
      (__attribute__((address_space(3))) i32x4*)l, 0, 0);
#else
  *(uint4*)l = *(const uint4*)g;
#endif
}

__device__ __forceinline__ void wait_async_copies() {
#if GEMM_ASYNC
#if __has_builtin(__builtin_amdgcn_s_wait_asynccnt)
  __builtin_amdgcn_s_wait_asynccnt(0);
#else
  asm volatile("s_wait_asynccnt 0" ::: "memory");
#endif
#endif
}

// ---------------- weight prep: fp32 -> zero-padded bf16 ----------------
__global__ void prep_weights_kernel(const float* __restrict__ fc1w, const float* __restrict__ fc1b,
                                    const float* __restrict__ fc2w, const float* __restrict__ fc2b,
                                    __bf16* w1, __bf16* w2, float* b1, float* b2) {
  int idx = blockIdx.x * blockDim.x + threadIdx.x;
  if (idx >= kFcP * kKP) return;
  int o = idx / kKP, k = idx % kKP;
  bool valid = (o < kFc) && (k < kPatch);
  float v1 = valid ? fc1w[o * kPatch + k] : 0.f;
  float v2 = valid ? fc2w[o * kPatch + k] : 0.f;
  w1[idx] = (__bf16)v1;
  w2[idx] = (__bf16)v2;
  if (k == 0) {
    b1[o] = (o < kFc) ? fc1b[o] : 0.f;
    b2[o] = (o < kFc) ? fc2b[o] : 0.f;
  }
}

// -------- front convs: b1 = conv3x3(b)+g_b, b2 = conv1x1(b)+theta_b,
// -------- written into pad-3 (134x134) planes, zero borders -----------
__global__ void conv_front_kernel(const float* __restrict__ b,
                                  const float* __restrict__ gw, const float* __restrict__ gb,
                                  const float* __restrict__ tw, const float* __restrict__ tb,
                                  float* __restrict__ b1p, float* __restrict__ b2p) {
  int idx = blockIdx.x * blockDim.x + threadIdx.x;
  if (idx >= kBatch * kCi * kPlane) return;
  int X = idx % kPW;
  int t = idx / kPW;
  int Y = t % kPW; t /= kPW;
  int co = t % kCi;
  int s  = t / kCi;
  int y = Y - 3, x = X - 3;
  float acc1 = 0.f, acc2 = 0.f;
  if (y >= 0 && y < kH && x >= 0 && x < kW) {
    acc1 = gb[co];
    acc2 = tb[co];
    const float* bp = b + ((size_t)(s * kCin) * kH + y) * kW + x;
    for (int ci = 0; ci < kCin; ++ci) {
      const float* bc = bp + (size_t)ci * kNPix;
      acc2 += bc[0] * tw[co * kCin + ci];
      const float* gwc = gw + ((co * kCin + ci) * 3) * 3;
      #pragma unroll
      for (int dy = -1; dy <= 1; ++dy) {
        int yy = y + dy;
        if (yy < 0 || yy >= kH) continue;
        #pragma unroll
        for (int dx = -1; dx <= 1; ++dx) {
          int xx = x + dx;
          if (xx < 0 || xx >= kW) continue;
          acc1 += bc[dy * kW + dx] * gwc[(dy + 1) * 3 + (dx + 1)];
        }
      }
    }
  }
  b1p[idx] = acc1;
  b2p[idx] = acc2;
}

// -------- soft_thr / soft_bias: 7x7 stride-4 convs over raw input ------
__global__ void soft_conv_kernel(const float* __restrict__ b,
                                 const float* __restrict__ tw, const float* __restrict__ tb,
                                 const float* __restrict__ bw, const float* __restrict__ bb,
                                 float* __restrict__ thr, float* __restrict__ bias_) {
  int idx = blockIdx.x * blockDim.x + threadIdx.x;
  if (idx >= kBatch * kNQ) return;
  int q = idx % kNQ, s = idx / kNQ;
  int qy = q / 32, qx = q % 32;
  float a1 = tb[0], a2 = bb[0];
  for (int ci = 0; ci < kCin; ++ci) {
    const float* bc = b + (size_t)(s * kCin + ci) * kNPix;
    for (int ky = 0; ky < kKs; ++ky) {
      int y = qy * 4 + ky - 1;               // pad top = 1
      if (y < 0 || y >= kH) continue;
      for (int kx = 0; kx < kKs; ++kx) {
        int x = qx * 4 + kx - 1;             // pad left = 1
        if (x < 0 || x >= kW) continue;
        float v = bc[y * kW + x];
        int wi_ = (ci * kKs + ky) * kKs + kx;
        a1 += v * tw[wi_];
        a2 += v * bw[wi_];
      }
    }
  }
  thr[idx]  = a1;
  bias_[idx] = a2;
}

// -------- patch matrix builders (row-major, K contiguous, bf16) --------
__global__ void build_p28_kernel(const float* __restrict__ b1p, __bf16* __restrict__ p28, int s) {
  int idx = blockIdx.x * blockDim.x + threadIdx.x;
  if (idx >= kNQ * kKP) return;
  int j = idx % kKP, q = idx / kKP;
  float v = 0.f;
  if (j < kPatch) {
    int c = j / 49, r = j % 49, ky = r / 7, kx = r % 7;
    int qy = q / 32, qx = q % 32;
    // stride-4 SAME pad is (1,2): pad3 coords offset by +2
    v = b1p[((size_t)(s * kCi + c) * kPW + (qy * 4 + ky + 2)) * kPW + (qx * 4 + kx + 2)];
  }
  p28[idx] = (__bf16)v;
}

__global__ void build_p112_2_kernel(const float* __restrict__ b1p, __bf16* __restrict__ p, int s) {
  int idx = blockIdx.x * blockDim.x + threadIdx.x;
  if (idx >= kNPix * kKP) return;
  int j = idx % kKP, n = idx / kKP;
  float v = 0.f;
  if (j < kPatch) {
    int c = j / 49, r = j % 49, ky = r / 7, kx = r % 7;
    int y = n / kW, x = n % kW;
    v = b1p[((size_t)(s * kCi + c) * kPW + (y + ky)) * kPW + (x + kx)];
  }
  p[idx] = (__bf16)v;
}

// p112 stored transposed: rows = patch feature o (784), cols = pixel n (16384)
__global__ void build_p112T_kernel(const float* __restrict__ b2p, __bf16* __restrict__ pT, int s) {
  int idx = blockIdx.x * blockDim.x + threadIdx.x;
  if (idx >= kPatch * kNPix) return;
  int n = idx % kNPix, o = idx / kNPix;
  int c = o / 49, r = o % 49, ky = r / 7, kx = r % 7;
  int y = n / kW, x = n % kW;
  float v = b2p[((size_t)(s * kCi + c) * kPW + (y + ky)) * kPW + (x + kx)];
  pT[idx] = (__bf16)v;
}

// ------------------- generic bf16 WMMA GEMM: C = A * B^T ---------------
// A: M x K row-major (bf16), B: N x K row-major (bf16). K % 32 == 0,
// M % 64 == 0 (all call sites). Block tile 64x128, wave tile 32x64,
// double-buffered LDS with async global->LDS prefetch.
// mode 0: Cf (fp32, ldc) = acc
// mode 1: Ch (bf16, ldc) = relu(acc + bias[n])
namespace {
constexpr int kLDT = 40;   // bf16 elems per LDS row (pad: conflict-free b128 frags)
constexpr int kBM  = 64;
constexpr int kBN  = 128;
}

__device__ __forceinline__ void gemm_tile_load(const __bf16* __restrict__ A,
                                               const __bf16* __restrict__ B,
                                               __bf16* sAbuf, __bf16* sBbuf,
                                               int bm, int bn, int k0,
                                               int N, int lda, int ldb, int tid) {
  // A tile: 64 rows x 32 bf16 (4 x 16B segs per row), no guard (M % 64 == 0)
  #pragma unroll
  for (int i = 0; i < 2; ++i) {
    int c = tid + i * 128;               // 0..255
    int row = c >> 2, seg = c & 3;
    copy16_g2l(A + (size_t)(bm + row) * lda + k0 + seg * 8,
               sAbuf + row * kLDT + seg * 8);
  }
  // B tile: 128 rows x 32 bf16; clamp row (tail cols are never stored)
  #pragma unroll
  for (int i = 0; i < 4; ++i) {
    int c = tid + i * 128;               // 0..511
    int row = c >> 2, seg = c & 3;
    int gn = bn + row;
    gn = gn < N ? gn : N - 1;
    copy16_g2l(B + (size_t)gn * ldb + k0 + seg * 8,
               sBbuf + row * kLDT + seg * 8);
  }
}

__launch_bounds__(128)
__global__ void gemm_bf16_kernel(const __bf16* __restrict__ A, const __bf16* __restrict__ B,
                                 float* __restrict__ Cf, __bf16* __restrict__ Ch,
                                 const float* __restrict__ bias,
                                 int M, int N, int K, int lda, int ldb, int ldc, int mode) {
  __shared__ __align__(16) __bf16 sA[2][kBM * kLDT];
  __shared__ __align__(16) __bf16 sB[2][kBN * kLDT];

  const int tid  = threadIdx.x;
  const int bm   = blockIdx.y * kBM;
  const int bn   = blockIdx.x * kBN;
  const int wave = tid >> 5;
  const int lane = tid & 31;
  const int wr   = (wave >> 1) * 32;            // wave M offset in block tile
  const int wc   = (wave & 1) * 64;             // wave N offset in block tile
  const int l    = lane & 15;
  const int hi   = lane >> 4;

  f32x8 acc[2][4] = {};

  gemm_tile_load(A, B, sA[0], sB[0], bm, bn, 0, N, lda, ldb, tid);
  wait_async_copies();
  __syncthreads();

  int buf = 0;
  for (int k0 = 0; k0 < K; k0 += 32) {
    if (k0 + 32 < K)
      gemm_tile_load(A, B, sA[buf ^ 1], sB[buf ^ 1], bm, bn, k0 + 32, N, lda, ldb, tid);

    // A frag (16x32): lane(l,hi) -> row l, K chunks [hi*8,+8) and [16+hi*8,+8)
    bf16x16 afr[2];
    #pragma unroll
    for (int i = 0; i < 2; ++i) {
      int r = wr + i * 16 + l;
      union { bf16x16 v; bf16x8 h[2]; } u;
      u.h[0] = *(const bf16x8*)(&sA[buf][r * kLDT + hi * 8]);
      u.h[1] = *(const bf16x8*)(&sA[buf][r * kLDT + 16 + hi * 8]);
      afr[i] = u.v;
    }
    // B frag (32x16) from B^T rows: lane(n,hi) -> row n, K chunks [hi*8,+8) and [16+hi*8,+8)
    bf16x16 bfr[4];
    #pragma unroll
    for (int j = 0; j < 4; ++j) {
      int r = wc + j * 16 + l;
      union { bf16x16 v; bf16x8 h[2]; } u;
      u.h[0] = *(const bf16x8*)(&sB[buf][r * kLDT + hi * 8]);
      u.h[1] = *(const bf16x8*)(&sB[buf][r * kLDT + 16 + hi * 8]);
      bfr[j] = u.v;
    }

    #pragma unroll
    for (int i = 0; i < 2; ++i)
      #pragma unroll
      for (int j = 0; j < 4; ++j)
        acc[i][j] = __builtin_amdgcn_wmma_f32_16x16x32_bf16(
            false, afr[i], false, bfr[j], (short)0, acc[i][j], false, false);

    wait_async_copies();
    __syncthreads();
    buf ^= 1;
  }

  // store: C 16x16 f32 frag -> row = hi*8 + r, col = l (rows always < M)
  #pragma unroll
  for (int i = 0; i < 2; ++i) {
    #pragma unroll
    for (int j = 0; j < 4; ++j) {
      #pragma unroll
      for (int r = 0; r < 8; ++r) {
        int row = bm + wr + i * 16 + hi * 8 + r;
        int col = bn + wc + j * 16 + l;
        if (col < N) {
          float v = acc[i][j][r];
          if (mode == 1) {
            v += bias[col];
            v = v > 0.f ? v : 0.f;
            Ch[(size_t)row * ldc + col] = (__bf16)v;
          } else {
            Cf[(size_t)row * ldc + col] = v;
          }
        }
      }
    }
  }
}

// ------- per-row mean / mask / softmax; score fp32 -> yi bf16 ----------
__launch_bounds__(256)
__global__ void softmax_mask_kernel(const float* __restrict__ score,
                                    const float* __restrict__ thrA,
                                    const float* __restrict__ biasA,
                                    __bf16* __restrict__ yi, int s) {
  __shared__ float rbuf[256];
  const int q   = blockIdx.x;
  const int tid = threadIdx.x;
  const float* sr = score + (size_t)q * kNPix;

  float sum = 0.f;
  for (int n = tid; n < kNPix; n += 256) sum += sr[n];
  rbuf[tid] = sum; __syncthreads();
  for (int o = 128; o > 0; o >>= 1) { if (tid < o) rbuf[tid] += rbuf[tid + o]; __syncthreads(); }
  const float mean = rbuf[0] / (float)kNPix;
  __syncthreads();

  const float t  = thrA[s * kNQ + q];
  const float bb = biasA[s * kNQ + q];

  float mx = 0.f;                                 // zero-logit entries always present
  for (int n = tid; n < kNPix; n += 256) {
    float sc = sr[n];
    float m  = sc - mean * t + bb; m = m > 0.f ? m : 0.f;
    float lg = sc * m * kScale;
    mx = fmaxf(mx, lg);
  }
  rbuf[tid] = mx; __syncthreads();
  for (int o = 128; o > 0; o >>= 1) { if (tid < o) rbuf[tid] = fmaxf(rbuf[tid], rbuf[tid + o]); __syncthreads(); }
  mx = rbuf[0]; __syncthreads();

  float ds = 0.f;
  for (int n = tid; n < kNPix; n += 256) {
    float sc = sr[n];
    float m  = sc - mean * t + bb; m = m > 0.f ? m : 0.f;
    float lg = sc * m * kScale;
    ds += __expf(lg - mx);
  }
  rbuf[tid] = ds; __syncthreads();
  for (int o = 128; o > 0; o >>= 1) { if (tid < o) rbuf[tid] += rbuf[tid + o]; __syncthreads(); }
  const float inv = 1.f / rbuf[0];

  for (int n = tid; n < kNPix; n += 256) {
    float sc = sr[n];
    float m  = sc - mean * t + bb; m = m > 0.f ? m : 0.f;
    float lg = sc * m * kScale;
    float y  = (m > 0.f) ? __expf(lg - mx) * inv : 0.f;
    yi[(size_t)q * kNPix + n] = (__bf16)y;
  }
}

// ------------- fold: overlap-add + count normalization ----------------
__global__ void fold_kernel(const float* __restrict__ agg, float* __restrict__ out, int s) {
  int idx = blockIdx.x * blockDim.x + threadIdx.x;
  if (idx >= kCi * kNPix) return;
  int x = idx % kW;
  int t = idx / kW;
  int y = t % kH;
  int c = t / kH;
  int Y = y + 3, X = x + 3;
  float acc = 0.f; int cnt = 0;
  #pragma unroll
  for (int ky = 0; ky < kKs; ++ky) {
    int ry = Y - ky;
    if (ry < 0 || (ry & 3)) continue;
    int qy = ry >> 2;
    if (qy >= 32) continue;
    #pragma unroll
    for (int kx = 0; kx < kKs; ++kx) {
      int rx = X - kx;
      if (rx < 0 || (rx & 3)) continue;
      int qx = rx >> 2;
      if (qx >= 32) continue;
      ++cnt;
      acc += agg[(size_t)(qy * 32 + qx) * kPatch + (c * 49 + ky * 7 + kx)];
    }
  }
  out[((size_t)(s * kCi + c) * kH + y) * kW + x] = acc / (float)(cnt ? cnt : 1);
}

// ----------------------------------------------------------------------
static inline size_t alignup(size_t x) { return (x + 255) & ~(size_t)255; }

extern "C" void kernel_launch(void* const* d_in, const int* in_sizes, int n_in,
                              void* d_out, int out_size, void* d_ws, size_t ws_size,
                              hipStream_t stream) {
  (void)in_sizes; (void)n_in; (void)out_size; (void)ws_size;
  const float* b      = (const float*)d_in[0];
  const float* g_w    = (const float*)d_in[1];
  const float* g_b    = (const float*)d_in[2];
  const float* th_w   = (const float*)d_in[3];
  const float* th_b   = (const float*)d_in[4];
  const float* fc1_w  = (const float*)d_in[5];
  const float* fc1_b  = (const float*)d_in[6];
  const float* fc2_w  = (const float*)d_in[7];
  const float* fc2_b  = (const float*)d_in[8];
  const float* thr_w  = (const float*)d_in[9];
  const float* thr_b  = (const float*)d_in[10];
  const float* bias_w = (const float*)d_in[11];
  const float* bias_b = (const float*)d_in[12];
  float* out = (float*)d_out;

  // workspace layout
  char* ws = (char*)d_ws;
  size_t off = 0;
  auto take = [&](size_t bytes) { size_t o = off; off = alignup(off + bytes); return o; };
  float*  b1p    = (float*)(ws + take((size_t)kBatch * kCi * kPlane * 4));
  float*  b2p    = (float*)(ws + take((size_t)kBatch * kCi * kPlane * 4));
  float*  thrA   = (float*)(ws + take((size_t)kBatch * kNQ * 4));
  float*  biasA  = (float*)(ws + take((size_t)kBatch * kNQ * 4));
  __bf16* fc1wp  = (__bf16*)(ws + take((size_t)kFcP * kKP * 2));
  __bf16* fc2wp  = (__bf16*)(ws + take((size_t)kFcP * kKP * 2));
  float*  fc1bp  = (float*)(ws + take((size_t)kFcP * 4));
  float*  fc2bp  = (float*)(ws + take((size_t)kFcP * 4));
  __bf16* p28    = (__bf16*)(ws + take((size_t)kNQ * kKP * 2));
  __bf16* p112_2 = (__bf16*)(ws + take((size_t)kNPix * kKP * 2));
  __bf16* p112T  = (__bf16*)(ws + take((size_t)kPatch * kNPix * 2));
  __bf16* wi     = (__bf16*)(ws + take((size_t)kNQ * kFcP * 2));
  __bf16* xi     = (__bf16*)(ws + take((size_t)kNPix * kFcP * 2));
  float*  score  = (float*)(ws + take((size_t)kNQ * kNPix * 4));
  __bf16* yi     = (__bf16*)(ws + take((size_t)kNQ * kNPix * 2));
  float*  aggb   = (float*)(ws + take((size_t)kNQ * kPatch * 4));

  const int T = 256;
  // weights + front convs + soft convs (whole batch)
  prep_weights_kernel<<<(kFcP * kKP + T - 1) / T, T, 0, stream>>>(
      fc1_w, fc1_b, fc2_w, fc2_b, fc1wp, fc2wp, fc1bp, fc2bp);
  conv_front_kernel<<<(kBatch * kCi * kPlane + T - 1) / T, T, 0, stream>>>(
      b, g_w, g_b, th_w, th_b, b1p, b2p);
  soft_conv_kernel<<<(kBatch * kNQ + T - 1) / T, T, 0, stream>>>(
      b, thr_w, thr_b, bias_w, bias_b, thrA, biasA);

  for (int s = 0; s < kBatch; ++s) {
    build_p28_kernel<<<(kNQ * kKP + T - 1) / T, T, 0, stream>>>(b1p, p28, s);
    build_p112_2_kernel<<<(kNPix * kKP + T - 1) / T, T, 0, stream>>>(b1p, p112_2, s);
    build_p112T_kernel<<<(kPatch * kNPix + T - 1) / T, T, 0, stream>>>(b2p, p112T, s);

    // wi = relu(p28 @ fc1^T + b1): M=1024, N=224, K=800
    gemm_bf16_kernel<<<dim3((kFcP + kBN - 1) / kBN, kNQ / kBM), 128, 0, stream>>>(
        p28, fc1wp, nullptr, wi, fc1bp, kNQ, kFcP, kKP, kKP, kKP, kFcP, 1);
    // xi = relu(p112_2 @ fc2^T + b2): M=16384, N=224, K=800
    gemm_bf16_kernel<<<dim3((kFcP + kBN - 1) / kBN, kNPix / kBM), 128, 0, stream>>>(
        p112_2, fc2wp, nullptr, xi, fc2bp, kNPix, kFcP, kKP, kKP, kKP, kFcP, 1);
    // score = wi @ xi^T: M=1024, N=16384, K=224
    gemm_bf16_kernel<<<dim3(kNPix / kBN, kNQ / kBM), 128, 0, stream>>>(
        wi, xi, score, nullptr, nullptr, kNQ, kNPix, kFcP, kFcP, kFcP, kNPix, 0);
    // mean / mask / softmax -> yi (bf16)
    softmax_mask_kernel<<<kNQ, 256, 0, stream>>>(score, thrA, biasA, yi, s);
    // agg = yi @ p112: M=1024, N=784, K=16384
    gemm_bf16_kernel<<<dim3((kPatch + kBN - 1) / kBN, kNQ / kBM), 128, 0, stream>>>(
        yi, p112T, aggb, nullptr, nullptr, kNQ, kPatch, kNPix, kNPix, kNPix, kPatch, 0);
    // overlap-add fold -> output
    fold_kernel<<<(kCi * kNPix + T - 1) / T, T, 0, stream>>>(aggb, out, s);
  }
}
